// AttentionMulti_12378095747549
// MI455X (gfx1250) — compile-verified, matrix operand order checked by reading, not run on hardware
//
#include <hip/hip_runtime.h>

typedef unsigned short u16;
typedef unsigned int u32;
typedef __attribute__((ext_vector_type(16))) __bf16 v16bf;
typedef __attribute__((ext_vector_type(8)))  __bf16 v8bf;
typedef __attribute__((ext_vector_type(8)))  float  v8f;

#define B_ 4
#define N_ 2048
#define E_ 1024
#define H_ 16
#define D_ 64
#define BH_ (B_ * H_)

__device__ __forceinline__ u16 f2bfu(float f) {
  union { float f; u32 u; } v; v.f = f;
  u32 r = v.u + 0x7fffu + ((v.u >> 16) & 1u);
  return (u16)(r >> 16);
}

__device__ __forceinline__ u32 pack2(float a, float b) {
  return (u32)f2bfu(a) | ((u32)f2bfu(b) << 16);
}

__device__ __forceinline__ v8f vzero8() {
  v8f z = {0.f, 0.f, 0.f, 0.f, 0.f, 0.f, 0.f, 0.f};
  return z;
}

// Load one 16x32 bf16 WMMA A/B fragment from a row-major [16+][stride] tile.
// ISA layout: lane&15 = row (A: M, B: N-column), kb=(lane>>4)*8,
// elements 0..7 -> K = kb..kb+7, elements 8..15 -> K = 16+kb .. 16+kb+7.
__device__ __forceinline__ v16bf load_frag(const u16* p, int stride, int lane) {
  int row = lane & 15;
  int kb  = (lane >> 4) << 3;
  const u16* q = p + (size_t)row * stride + kb;
  v8bf lo = *(const v8bf*)(q);
  v8bf hi = *(const v8bf*)(q + 16);
  return __builtin_shufflevector(lo, hi, 0,1,2,3,4,5,6,7,8,9,10,11,12,13,14,15);
}

__device__ __forceinline__ v8f wmma_bf16(v16bf a, v16bf b, v8f c) {
  return __builtin_amdgcn_wmma_f32_16x16x32_bf16(
      false, a, false, b, (short)0, c, false, false);
}

// ---------------------------------------------------------------------------
// Streaming fp32 -> bf16 (8 elements / thread)
// ---------------------------------------------------------------------------
__global__ __launch_bounds__(256) void convert_f32_to_bf16(
    const float* __restrict__ in, u16* __restrict__ out, long long n) {
  long long i = ((long long)blockIdx.x * 256 + threadIdx.x) * 8;
  if (i >= n) return;
  float4 f0 = *(const float4*)(in + i);
  float4 f1 = *(const float4*)(in + i + 4);
  uint4 w;
  w.x = pack2(f0.x, f0.y);
  w.y = pack2(f0.z, f0.w);
  w.z = pack2(f1.x, f1.y);
  w.w = pack2(f1.z, f1.w);
  *(uint4*)(out + i) = w;
}

// ---------------------------------------------------------------------------
// Transpose-convert fp32 [K][N] -> bf16 [N][K]
// ---------------------------------------------------------------------------
__global__ __launch_bounds__(256) void transpose_f32_to_bf16(
    const float* __restrict__ in, u16* __restrict__ out, int K, int N) {
  long long idx = (long long)blockIdx.x * 256 + threadIdx.x;
  if (idx >= (long long)K * N) return;
  int n = (int)(idx / K);
  int k = (int)(idx % K);
  out[idx] = f2bfu(in[(size_t)k * N + n]);
}

// ---------------------------------------------------------------------------
// QKV GEMM: C[8192][3072] = xb[8192][1024] (bf16) * wqkv_t[3072][1024]^T
//           + bias, scattered into Q[BH][N][64], K[BH][N][64], Vt[BH][64][N].
// Block tile 128x128, BK=64, 8 waves each 32(M) x 64(N).
// sched_barrier(0) splits the fill into "issue all 8 global loads" then
// "store to LDS" so the loads stay batched in distinct registers.
// ---------------------------------------------------------------------------
__global__ __launch_bounds__(256) void gemm_qkv_kernel(
    const u16* __restrict__ xb, const u16* __restrict__ wt,
    const float* __restrict__ bias,
    u16* __restrict__ qbuf, u16* __restrict__ kbuf, u16* __restrict__ vtbuf) {
  __shared__ u16 As[128 * 72];
  __shared__ u16 Bs[128 * 72];

  const int tid  = threadIdx.x;
  const int lane = tid & 31;
  const int wave = tid >> 5;
  const int wm = wave & 3;   // 0..3 -> 32-row slab
  const int wn = wave >> 2;  // 0..1 -> 64-col slab
  const int n0 = blockIdx.x * 128;
  const int m0 = blockIdx.y * 128;

  v8f acc[2][4];
#pragma unroll
  for (int i = 0; i < 2; ++i)
#pragma unroll
    for (int j = 0; j < 4; ++j) acc[i][j] = vzero8();

  const int lrow  = tid >> 1;
  const int lhalf = tid & 1;
  const u16* aRow = xb + (size_t)(m0 + lrow) * E_ + lhalf * 32;
  const u16* bRow = wt + (size_t)(n0 + lrow) * E_ + lhalf * 32;
  uint4* ad = (uint4*)(As + lrow * 72 + lhalf * 32);
  uint4* bd = (uint4*)(Bs + lrow * 72 + lhalf * 32);

  for (int k0 = 0; k0 < E_; k0 += 64) {
    // batched fill: all 8 loads issue before any LDS store
    uint4 at[4], bt[4];
#pragma unroll
    for (int i = 0; i < 4; ++i) {
      at[i] = ((const uint4*)(aRow + k0))[i];
      bt[i] = ((const uint4*)(bRow + k0))[i];
    }
    __builtin_amdgcn_sched_barrier(0);   // keep loads batched (distinct regs)
#pragma unroll
    for (int i = 0; i < 4; ++i) { ad[i] = at[i]; bd[i] = bt[i]; }
    __syncthreads();
    // warm the cache for the next tile; overlaps with the WMMAs below
    if (k0 + 64 < E_) {
      __builtin_prefetch(aRow + k0 + 64, 0, 3);
      __builtin_prefetch(bRow + k0 + 64, 0, 3);
    }
#pragma unroll
    for (int ks = 0; ks < 2; ++ks) {
      v16bf af[2], bfv[4];
#pragma unroll
      for (int i = 0; i < 2; ++i)
        af[i] = load_frag(As + (wm * 32 + i * 16) * 72 + ks * 32, 72, lane);
#pragma unroll
      for (int j = 0; j < 4; ++j)
        bfv[j] = load_frag(Bs + (wn * 64 + j * 16) * 72 + ks * 32, 72, lane);
#pragma unroll
      for (int i = 0; i < 2; ++i)
#pragma unroll
        for (int j = 0; j < 4; ++j)
          acc[i][j] = wmma_bf16(af[i], bfv[j], acc[i][j]);
    }
    __syncthreads();
  }

  // Epilogue: + bias, scatter to Q / K / Vt (bf16)
  const int colHalf = lane >> 4;
  const int nlo     = lane & 15;
#pragma unroll
  for (int j = 0; j < 4; ++j) {
    int nGlob = n0 + wn * 64 + j * 16 + nlo;
    float bv = bias[nGlob];
    int which = nGlob >> 10;        // 0=Q 1=K 2=V
    int rem   = nGlob & 1023;
    int h = rem >> 6, d = rem & 63;
#pragma unroll
    for (int i = 0; i < 2; ++i) {
#pragma unroll
      for (int r = 0; r < 8; ++r) {
        int mGlob = m0 + wm * 32 + i * 16 + r + (colHalf << 3);
        int bq  = mGlob >> 11;          // batch
        int row = mGlob & (N_ - 1);
        u16 val = f2bfu(acc[i][j][r] + bv);
        size_t head = (size_t)(bq * H_ + h);
        if (which == 0)      qbuf[(head * N_ + row) * D_ + d] = val;
        else if (which == 1) kbuf[(head * N_ + row) * D_ + d] = val;
        else                 vtbuf[(head * D_ + d) * N_ + row] = val;
      }
    }
  }
}

// ---------------------------------------------------------------------------
// Flash attention. Block = 8 waves x 16 q-rows = 128 q-rows; K/Vt tiles for
// each 32-key step are staged once in LDS and shared by all 8 waves (8x less
// global/L2 traffic than per-wave loads); next tile warmed via prefetch.
// ---------------------------------------------------------------------------
__global__ __launch_bounds__(256) void attn_kernel(
    const u16* __restrict__ Q, const u16* __restrict__ K,
    const u16* __restrict__ VT, const int* __restrict__ mask,
    u16* __restrict__ Oout) {
  __shared__ u16 Ks[32 * 72];        // [key][d]   (4.5 KB)
  __shared__ u16 Vs[64 * 40];        // [d][key]   (5 KB)
  __shared__ u16 plds[8][16 * 40];   // per-wave P tile (10 KB)

  const int tid  = threadIdx.x;
  const int lane = tid & 31;
  const int wave = tid >> 5;
  const int bh = blockIdx.y;
  const int bq = bh >> 4;
  const int h  = bh & 15;
  const int qRow0 = blockIdx.x * 128 + wave * 16;

  const u16* Qh  = Q  + (size_t)bh * N_ * D_;
  const u16* Kh  = K  + (size_t)bh * N_ * D_;
  const u16* VTh = VT + (size_t)bh * D_ * N_;
  const int* maskB = mask + bq * N_;
  u16* myP = &plds[wave][0];

  const int colHalf = lane >> 4;
  const int nlo     = lane & 15;

  // Cooperative staging indices: K tile 32x64, V tile 64x32, 16B per thread.
  const int krow = tid >> 3, kcol = (tid & 7) * 8;
  const int vrow = tid >> 2, vcol = (tid & 3) * 8;

  // Q fragments (16 rows x 64 d) live in registers for the whole pass.
  v16bf qf[2];
#pragma unroll
  for (int j = 0; j < 2; ++j)
    qf[j] = load_frag(Qh + (size_t)qRow0 * D_ + j * 32, D_, lane);

  v8f o[4];
#pragma unroll
  for (int t = 0; t < 4; ++t) o[t] = vzero8();
  float m_i[8], l_i[8];
#pragma unroll
  for (int r = 0; r < 8; ++r) { m_i[r] = -1e30f; l_i[r] = 0.f; }

  for (int kt = 0; kt < N_; kt += 32) {
    // stage current K/V tile into LDS (both loads issued before stores)
    {
      uint4 kreg = *(const uint4*)(Kh + (size_t)(kt + krow) * D_ + kcol);
      uint4 vreg = *(const uint4*)(VTh + (size_t)vrow * N_ + kt + vcol);
      __builtin_amdgcn_sched_barrier(0);
      *(uint4*)(Ks + krow * 72 + kcol) = kreg;
      *(uint4*)(Vs + vrow * 40 + vcol) = vreg;
    }
    __syncthreads();
    // warm the cache for the next tile; overlaps with compute below
    int ktn = kt + 32;
    if (ktn < N_) {
      __builtin_prefetch(Kh + (size_t)(ktn + krow) * D_ + kcol, 0, 3);
      __builtin_prefetch(VTh + (size_t)vrow * N_ + ktn + vcol, 0, 3);
    }

    // S = Q * K^T  (two 16x16 key tiles, reduction D=64 in two 32-steps)
    v8f s0 = vzero8(), s1 = vzero8();
#pragma unroll
    for (int ks = 0; ks < 2; ++ks) {
      v16bf kf0 = load_frag(Ks + ks * 32, 72, lane);
      v16bf kf1 = load_frag(Ks + 16 * 72 + ks * 32, 72, lane);
      s0 = wmma_bf16(qf[ks], kf0, s0);
      s1 = wmma_bf16(qf[ks], kf1, s1);
    }

    const int ok0 = maskB[kt + nlo];
    const int ok1 = maskB[kt + 16 + nlo];

    // Online softmax: lane holds rows M = r + 8*(lane>>4); a row's 16 values
    // live in one 16-lane half -> xor-shuffle reduce with masks 1,2,4,8.
#pragma unroll
    for (int r = 0; r < 8; ++r) {
      float x0 = ok0 ? s0[r] * 0.5f : -1e30f;   // scale = 1/sqrt(4)
      float x1 = ok1 ? s1[r] * 0.5f : -1e30f;
      float mx = fmaxf(x0, x1);
#pragma unroll
      for (int d = 1; d < 16; d <<= 1) mx = fmaxf(mx, __shfl_xor(mx, d, 32));
      float mnew = fmaxf(m_i[r], mx);
      float sc = __expf(m_i[r] - mnew);
      m_i[r] = mnew;
      float p0 = __expf(x0 - mnew);
      float p1 = __expf(x1 - mnew);
      float ps = p0 + p1;
#pragma unroll
      for (int d = 1; d < 16; d <<= 1) ps += __shfl_xor(ps, d, 32);
      l_i[r] = l_i[r] * sc + ps;
      o[0][r] *= sc; o[1][r] *= sc; o[2][r] *= sc; o[3][r] *= sc;
      int prow = r + (colHalf << 3);
      myP[prow * 40 + nlo]      = f2bfu(p0);
      myP[prow * 40 + 16 + nlo] = f2bfu(p1);
    }
    __builtin_amdgcn_wave_barrier();   // keep LDS C->A relayout ordered

    // O += P * V  (B-fragments contiguous thanks to Vt layout)
    v16bf pf = load_frag(myP, 40, lane);
#pragma unroll
    for (int t = 0; t < 4; ++t) {
      v16bf vf = load_frag(Vs + (t * 16) * 40, 40, lane);
      o[t] = wmma_bf16(pf, vf, o[t]);
    }
    __syncthreads();   // all waves done with Ks/Vs before next commit
  }

  // Normalize and store bf16 into attn_out [B][N][H*64]
#pragma unroll
  for (int r = 0; r < 8; ++r) {
    float inv = 1.0f / l_i[r];
    int row = qRow0 + r + (colHalf << 3);
    size_t rb = ((size_t)(bq * N_ + row)) * (H_ * D_) + h * D_;
#pragma unroll
    for (int t = 0; t < 4; ++t)
      Oout[rb + t * 16 + nlo] = f2bfu(o[t][r] * inv);
  }
}

// ---------------------------------------------------------------------------
// Output GEMM: out[8192][1024] = attn_out[8192][1024](bf16) *
//              wout_t[1024][1024]^T (bf16) + b_out, fp32 result.
// ---------------------------------------------------------------------------
__global__ __launch_bounds__(256) void gemm_out_kernel(
    const u16* __restrict__ a, const u16* __restrict__ wt,
    const float* __restrict__ bias, float* __restrict__ out) {
  __shared__ u16 As[128 * 72];
  __shared__ u16 Bs[128 * 72];

  const int tid  = threadIdx.x;
  const int lane = tid & 31;
  const int wave = tid >> 5;
  const int wm = wave & 3;
  const int wn = wave >> 2;
  const int n0 = blockIdx.x * 128;
  const int m0 = blockIdx.y * 128;

  v8f acc[2][4];
#pragma unroll
  for (int i = 0; i < 2; ++i)
#pragma unroll
    for (int j = 0; j < 4; ++j) acc[i][j] = vzero8();

  const int lrow  = tid >> 1;
  const int lhalf = tid & 1;
  const u16* aRow = a  + (size_t)(m0 + lrow) * 1024 + lhalf * 32;
  const u16* bRow = wt + (size_t)(n0 + lrow) * 1024 + lhalf * 32;
  uint4* ad = (uint4*)(As + lrow * 72 + lhalf * 32);
  uint4* bd = (uint4*)(Bs + lrow * 72 + lhalf * 32);

  for (int k0 = 0; k0 < 1024; k0 += 64) {
    uint4 at[4], bt[4];
#pragma unroll
    for (int i = 0; i < 4; ++i) {
      at[i] = ((const uint4*)(aRow + k0))[i];
      bt[i] = ((const uint4*)(bRow + k0))[i];
    }
    __builtin_amdgcn_sched_barrier(0);   // keep loads batched (distinct regs)
#pragma unroll
    for (int i = 0; i < 4; ++i) { ad[i] = at[i]; bd[i] = bt[i]; }
    __syncthreads();
    if (k0 + 64 < 1024) {
      __builtin_prefetch(aRow + k0 + 64, 0, 3);
      __builtin_prefetch(bRow + k0 + 64, 0, 3);
    }
#pragma unroll
    for (int ks = 0; ks < 2; ++ks) {
      v16bf af[2], bfv[4];
#pragma unroll
      for (int i = 0; i < 2; ++i)
        af[i] = load_frag(As + (wm * 32 + i * 16) * 72 + ks * 32, 72, lane);
#pragma unroll
      for (int j = 0; j < 4; ++j)
        bfv[j] = load_frag(Bs + (wn * 64 + j * 16) * 72 + ks * 32, 72, lane);
#pragma unroll
      for (int i = 0; i < 2; ++i)
#pragma unroll
        for (int j = 0; j < 4; ++j)
          acc[i][j] = wmma_bf16(af[i], bfv[j], acc[i][j]);
    }
    __syncthreads();
  }

  const int colHalf = lane >> 4;
  const int nlo     = lane & 15;
#pragma unroll
  for (int j = 0; j < 4; ++j) {
    int nGlob = n0 + wn * 64 + j * 16 + nlo;
    float bv = bias[nGlob];
#pragma unroll
    for (int i = 0; i < 2; ++i) {
#pragma unroll
      for (int r = 0; r < 8; ++r) {
        int mGlob = m0 + wm * 32 + i * 16 + r + (colHalf << 3);
        out[(size_t)mGlob * 1024 + nGlob] = acc[i][j][r] + bv;
      }
    }
  }
}

// ---------------------------------------------------------------------------
extern "C" void kernel_launch(void* const* d_in, const int* in_sizes, int n_in,
                              void* d_out, int out_size, void* d_ws, size_t ws_size,
                              hipStream_t stream) {
  const float* x     = (const float*)d_in[0];
  const int*   mask  = (const int*)d_in[1];
  const float* w_qkv = (const float*)d_in[2];
  const float* b_qkv = (const float*)d_in[3];
  const float* w_out = (const float*)d_in[4];
  const float* b_out = (const float*)d_in[5];
  float* out = (float*)d_out;

  char* ws = (char*)d_ws;
  u16* wqkv_t  = (u16*)ws;  ws += (size_t)3072 * 1024 * 2;
  u16* wout_t  = (u16*)ws;  ws += (size_t)1024 * 1024 * 2;
  u16* qbuf    = (u16*)ws;  ws += (size_t)BH_ * N_ * D_ * 2;
  u16* kbuf    = (u16*)ws;  ws += (size_t)BH_ * N_ * D_ * 2;
  u16* vtbuf   = (u16*)ws;  ws += (size_t)BH_ * N_ * D_ * 2;
  // xb (bf16 copy of x) is consumed by gemm_qkv before attn writes attn_out:
  // alias the two 16 MB regions to keep workspace at 72 MB.
  u16* xb      = (u16*)ws;
  u16* attnout = (u16*)ws;  ws += (size_t)B_ * N_ * H_ * D_ * 2;

  const long long nx = (long long)B_ * N_ * E_;
  convert_f32_to_bf16<<<(unsigned)(nx / (256 * 8)), 256, 0, stream>>>(x, xb, nx);
  transpose_f32_to_bf16<<<(3072 * 1024) / 256, 256, 0, stream>>>(w_qkv, wqkv_t, 1024, 3072);
  transpose_f32_to_bf16<<<(1024 * 1024) / 256, 256, 0, stream>>>(w_out, wout_t, 1024, 1024);

  gemm_qkv_kernel<<<dim3(3072 / 128, (B_ * N_) / 128), 256, 0, stream>>>(
      xb, wqkv_t, b_qkv, qbuf, kbuf, vtbuf);

  attn_kernel<<<dim3(N_ / 128, BH_), 256, 0, stream>>>(
      qbuf, kbuf, vtbuf, mask, attnout);

  gemm_out_kernel<<<dim3(1024 / 128, (B_ * N_) / 128), 256, 0, stream>>>(
      attnout, wout_t, b_out, out);
}
